// MultiScaleAttention_37864431681670
// MI455X (gfx1250) — matrix operand order repaired and checked by reading, not compiled, not run on hardware
//
#include <hip/hip_runtime.h>
#include <hip/hip_bf16.h>

// Problem constants (from reference): B=2, S=2048, H=1024, NH=16, HD=64, MAX_SCALE=4
#define S_LEN  2048
#define H_DIM  1024
#define NHEAD  16
#define HDIM   64
#define BATCH  2
#define NSCALE 4

typedef __attribute__((ext_vector_type(16))) _Float16 v16h;
typedef __attribute__((ext_vector_type(8)))  float    v8f;
typedef __attribute__((ext_vector_type(2)))  __fp16   pk2h;   // cvt_pkrtz result type
typedef __attribute__((ext_vector_type(4)))  unsigned v4u;
typedef __attribute__((ext_vector_type(8)))  unsigned v8u;

union V16HU { uint4 u[2]; v16h v; };
union H8U   { uint4 u; _Float16 h[8]; };
union PK2U  { pk2h h2[2]; uint2 u; };

__device__ __forceinline__ v8f wmma_f16(v16h a, v16h b, v8f c) {
  // D = A(16x32 f16) x B(32x16 f16) + C(16x16 f32)
  return __builtin_amdgcn_wmma_f32_16x16x32_f16(false, a, false, b, (short)0, c,
                                                false, false);
}

// Build a 16-half fragment from two 16-byte chunks (global or LDS).
__device__ __forceinline__ v16h make_frag(const _Float16* p0, const _Float16* p1) {
  V16HU t;
  t.u[0] = *(const uint4*)p0;
  t.u[1] = *(const uint4*)p1;
  return t.v;
}

// ---------------------------------------------------------------------------
// Kernel A: qkv = x @ Wqkv + bqkv, scattered to q/k/v f16 in (B, NH, S, HD)
// M = B*S = 4096, N = 3072, K = 1024. WG: 128x128 tile, 8 waves (2x4).
// ---------------------------------------------------------------------------
__global__ __launch_bounds__(256) void qkv_proj_kernel(
    const float* __restrict__ x, const float* __restrict__ Wqkv,
    const float* __restrict__ bqkv,
    _Float16* __restrict__ qf, _Float16* __restrict__ kf,
    _Float16* __restrict__ vf)
{
  __shared__ _Float16 As[128][40];   // [m][k], pad to 80B rows (16B aligned)
  __shared__ _Float16 Bs[128][40];   // [n][k] (W transposed for B-frags)
  const int m0 = blockIdx.x * 128;
  const int n0 = blockIdx.y * 128;
  const int tid  = threadIdx.x;
  const int wave = tid >> 5, lane = tid & 31;
  const int halfw = lane >> 4, r = lane & 15;
  const int wm = wave >> 2, wn = wave & 3;   // 2 (M) x 4 (N) waves

  v8f acc[4][2] = {};

  for (int k0 = 0; k0 < H_DIM; k0 += 32) {
    __syncthreads();
    // A tile: 128x32 f32 -> f16 LDS (float4 load, packed v_cvt_pk_f16_f32)
    #pragma unroll
    for (int i = 0; i < 4; ++i) {
      int chunk = tid + i * 256;          // 1024 float4 chunks
      int row = chunk >> 3;
      int kk  = (chunk & 7) * 4;
      float4 xv = *(const float4*)(x + (size_t)(m0 + row) * H_DIM + k0 + kk);
      PK2U pk;
      pk.h2[0] = __builtin_amdgcn_cvt_pkrtz(xv.x, xv.y);
      pk.h2[1] = __builtin_amdgcn_cvt_pkrtz(xv.z, xv.w);
      *(uint2*)(&As[row][kk]) = pk.u;     // one ds_store_b64
    }
    // B tile: 32x128 f32, stored transposed [n][k]
    #pragma unroll
    for (int i = 0; i < 4; ++i) {
      int chunk = tid + i * 256;
      int krow = chunk >> 5;
      int nn   = (chunk & 31) * 4;
      float4 wv = *(const float4*)(Wqkv + (size_t)(k0 + krow) * (3*H_DIM) + n0 + nn);
      Bs[nn+0][krow] = (_Float16)wv.x;
      Bs[nn+1][krow] = (_Float16)wv.y;
      Bs[nn+2][krow] = (_Float16)wv.z;
      Bs[nn+3][krow] = (_Float16)wv.w;
    }
    __syncthreads();

    v16h af[4], bf[2];
    #pragma unroll
    for (int mt = 0; mt < 4; ++mt) {
      const _Float16* ap = &As[wm*64 + mt*16 + r][0];
      af[mt] = make_frag(ap + 8*halfw, ap + 16 + 8*halfw);   // ISA A-frag layout
    }
    #pragma unroll
    for (int nt = 0; nt < 2; ++nt) {
      const _Float16* bp = &Bs[wn*32 + nt*16 + r][0];
      bf[nt] = make_frag(bp + 8*halfw, bp + 16 + 8*halfw);  // ISA B-frag layout
    }
    #pragma unroll
    for (int mt = 0; mt < 4; ++mt)
      #pragma unroll
      for (int nt = 0; nt < 2; ++nt)
        acc[mt][nt] = wmma_f16(af[mt], bf[nt], acc[mt][nt]);
  }

  // Epilogue: + bias, scatter into q/k/v (B, NH, S, HD) as f16
  #pragma unroll
  for (int nt = 0; nt < 2; ++nt) {
    int n_g   = n0 + wn*32 + nt*16 + r;
    int which = n_g >> 10;
    int hn    = n_g & 1023;
    int hh = hn >> 6, dd = hn & 63;
    _Float16* dst = (which == 0) ? qf : (which == 1) ? kf : vf;
    float bias = bqkv[n_g];
    #pragma unroll
    for (int mt = 0; mt < 4; ++mt) {
      #pragma unroll
      for (int g = 0; g < 8; ++g) {
        int m_g = m0 + wm*64 + mt*16 + 8*halfw + g;   // C-layout row
        int bb = m_g >> 11, ss = m_g & 2047;
        dst[(((size_t)bb*NHEAD + hh)*S_LEN + ss)*HDIM + dd] =
            (_Float16)(acc[mt][nt][g] + bias);
      }
    }
  }
}

// ---------------------------------------------------------------------------
// Kernel C: scale_weights = softmax(x @ Wscale + bscale)   (M=4096, K=1024, N=4)
// One wave per row.
// ---------------------------------------------------------------------------
__global__ __launch_bounds__(256) void scale_softmax_kernel(
    const float* __restrict__ x, const float* __restrict__ Wscale,
    const float* __restrict__ bscale, float* __restrict__ sw)
{
  int row  = blockIdx.x * 8 + (threadIdx.x >> 5);
  int lane = threadIdx.x & 31;
  float a0 = 0.f, a1 = 0.f, a2 = 0.f, a3 = 0.f;
  for (int kk = lane; kk < H_DIM; kk += 32) {
    float  xv = x[(size_t)row * H_DIM + kk];
    float4 wv = *(const float4*)(Wscale + (size_t)kk * 4);
    a0 += xv * wv.x; a1 += xv * wv.y; a2 += xv * wv.z; a3 += xv * wv.w;
  }
  #pragma unroll
  for (int off = 16; off >= 1; off >>= 1) {
    a0 += __shfl_xor(a0, off, 32);
    a1 += __shfl_xor(a1, off, 32);
    a2 += __shfl_xor(a2, off, 32);
    a3 += __shfl_xor(a3, off, 32);
  }
  a0 += bscale[0]; a1 += bscale[1]; a2 += bscale[2]; a3 += bscale[3];
  float mx = fmaxf(fmaxf(a0, a1), fmaxf(a2, a3));
  float e0 = __expf(a0 - mx), e1 = __expf(a1 - mx);
  float e2 = __expf(a2 - mx), e3 = __expf(a3 - mx);
  float inv = 1.0f / (e0 + e1 + e2 + e3);
  if (lane == 0) {
    float4 o; o.x = e0*inv; o.y = e1*inv; o.z = e2*inv; o.w = e3*inv;
    *(float4*)(sw + (size_t)row * 4) = o;
  }
}

// ---------------------------------------------------------------------------
// Kernel B: multi-scale flash attention.
// WG = (b, h, 128-query block); 8 waves x 16 query rows each.
// K tile staged by the Tensor Data Mover (tensor_load_to_lds), V transposed
// through VGPRs. Loops scales internally, folding w/l-weighted outputs.
// ---------------------------------------------------------------------------
__global__ __launch_bounds__(256) void msattn_kernel(
    const _Float16* __restrict__ qf, const _Float16* __restrict__ kf,
    const _Float16* __restrict__ vf, const float* __restrict__ sw,
    _Float16* __restrict__ attn)
{
  __shared__ _Float16 Klds[64][72];        // [key][d], 16B row pad (TDM pads)
  __shared__ _Float16 Vlds[64][72];        // [d][key] (transposed)
  __shared__ _Float16 Plds[8][16][72];     // per-wave P staging [qrow][key]

  const int bh = blockIdx.x;               // b*16 + h
  const int b  = bh >> 4, h = bh & 15;
  const int q0 = blockIdx.y * 128;
  const int tid  = threadIdx.x;
  const int wave = tid >> 5, lane = tid & 31;
  const int halfw = lane >> 4, r = lane & 15;
  const int qrow = q0 + wave * 16;
  const float inv_sqrt = 0.125f;           // 1/sqrt(HD=64)

  const _Float16* qbase = qf + ((size_t)bh * S_LEN + qrow) * HDIM;
  const _Float16* kbase = kf + (size_t)bh * S_LEN * HDIM;
  const _Float16* vbase = vf + (size_t)bh * S_LEN * HDIM;

  // Resident Q fragments (16 rows x 64 d = 2 A-frags)
  v16h qfr[2];
  #pragma unroll
  for (int s = 0; s < 2; ++s) {
    const _Float16* qp = qbase + (size_t)r * HDIM + s*32;
    qfr[s] = make_frag(qp + 8*halfw, qp + 16 + 8*halfw);
  }

  float ofin[4][8] = {};

  for (int sc = 0; sc < NSCALE; ++sc) {
    const int dil = 1 << sc;
    const int Sk  = S_LEN >> sc;
    v8f os[4] = {};
    float mrow[8], lrow[8];
    #pragma unroll
    for (int g = 0; g < 8; ++g) { mrow[g] = -1e30f; lrow[g] = 0.f; }

    for (int j0 = 0; j0 < Sk; j0 += 64) {
      __syncthreads();

      // --- K tile via Tensor Data Mover: 64 dilated rows x 64 f16 -> Klds.
      // D# group0: count=1 | lds_addr | global_addr | type=2.
      // D# group1: data_size=2B, pad_enable, pad every 32 DWORDs (one 128B
      // row) by 4 DWORDs (16B) to produce the [64][72] padded layout;
      // tensor_dim0=tile_dim0=64, tensor_dim1=tile_dim1=64,
      // tensor_dim0_stride = dil*64 elements (the dilation).
      if (wave == 0) {
        unsigned long long ga =
            (unsigned long long)(uintptr_t)(kbase + (size_t)(j0) * dil * HDIM);
        unsigned lds_off = (unsigned)(uintptr_t)(&Klds[0][0]);
        v4u g0;
        g0.x = 1u;                                   // count = 1 (valid D#)
        g0.y = lds_off;                              // lds_addr (bytes)
        g0.z = (unsigned)ga;                         // global_addr[31:0]
        g0.w = ((unsigned)(ga >> 32) & 0x01FFFFFFu)  // global_addr[56:32]
               | 0x80000000u;                        // type = 2 ("image")
        v8u g1;
        g1[0] = (1u << 16)        // data_size = 2 bytes
              | (1u << 20)        // pad_enable
              | (4u << 22)        // pad_interval: 32 DWORDs
              | (3u << 25);       // pad_amount:   4 DWORDs
        g1[1] = 64u << 16;        // tensor_dim0 = 64
        g1[2] = 64u << 16;        // tensor_dim1[15:0] = 64
        g1[3] = 64u << 16;        // tile_dim0 = 64
        g1[4] = 64u;              // tile_dim1 = 64 (tile_dim2 = 0)
        g1[5] = (unsigned)(dil * HDIM);  // tensor_dim0_stride (elements)
        g1[6] = 0u;
        g1[7] = 0u;
        asm volatile("tensor_load_to_lds %0, %1" :: "s"(g0), "s"(g1) : "memory");
      }

      // --- V tile through VGPRs, transposed on store: Vlds[d][key]
      #pragma unroll
      for (int i = 0; i < 2; ++i) {
        int chunk = tid + i * 256;         // 512 chunks of 8 halves
        int j  = chunk >> 3;
        int cc = (chunk & 7) * 8;
        H8U tv;
        tv.u = *(const uint4*)(vbase + (size_t)(j0 + j) * dil * HDIM + cc);
        #pragma unroll
        for (int e = 0; e < 8; ++e) Vlds[cc + e][j] = tv.h[e];
      }
      // Prefetch next key block into cache (global_prefetch_b8)
      if (j0 + 64 < Sk) {
        int j = tid & 63;
        __builtin_prefetch(kbase + (size_t)(j0 + 64 + j) * dil * HDIM, 0, 0);
        __builtin_prefetch(vbase + (size_t)(j0 + 64 + j) * dil * HDIM, 0, 0);
      }
      if (wave == 0) __builtin_amdgcn_s_wait_tensorcnt(0);
      __syncthreads();

      // S = Q K^T  (16 x 64 per wave)
      v8f st[4] = {};
      #pragma unroll
      for (int s = 0; s < 2; ++s) {
        #pragma unroll
        for (int t = 0; t < 4; ++t) {
          const _Float16* kp = &Klds[t*16 + r][s*32 + 8*halfw];
          st[t] = wmma_f16(qfr[s], make_frag(kp, kp + 16), st[t]);
        }
      }
      #pragma unroll
      for (int t = 0; t < 4; ++t)
        #pragma unroll
        for (int g = 0; g < 8; ++g) st[t][g] *= inv_sqrt;

      // Online softmax: row = 8*halfw + g, cols striped over 16 lanes
      float mx[8];
      #pragma unroll
      for (int g = 0; g < 8; ++g)
        mx[g] = fmaxf(fmaxf(st[0][g], st[1][g]), fmaxf(st[2][g], st[3][g]));
      #pragma unroll
      for (int off = 1; off < 16; off <<= 1)
        #pragma unroll
        for (int g = 0; g < 8; ++g)
          mx[g] = fmaxf(mx[g], __shfl_xor(mx[g], off, 32));

      float corr[8], rs[8];
      #pragma unroll
      for (int g = 0; g < 8; ++g) {
        float nm = fmaxf(mrow[g], mx[g]);
        corr[g] = __expf(mrow[g] - nm);
        mrow[g] = nm;
        rs[g] = 0.f;
      }
      #pragma unroll
      for (int t = 0; t < 4; ++t) {
        #pragma unroll
        for (int g = 0; g < 8; ++g) {
          float p = __expf(st[t][g] - mrow[g]);
          rs[g] += p;
          Plds[wave][8*halfw + g][t*16 + r] = (_Float16)p;
        }
      }
      #pragma unroll
      for (int off = 1; off < 16; off <<= 1)
        #pragma unroll
        for (int g = 0; g < 8; ++g)
          rs[g] += __shfl_xor(rs[g], off, 32);
      #pragma unroll
      for (int g = 0; g < 8; ++g) lrow[g] = lrow[g] * corr[g] + rs[g];
      #pragma unroll
      for (int t = 0; t < 4; ++t)
        #pragma unroll
        for (int g = 0; g < 8; ++g) os[t][g] *= corr[g];

      __builtin_amdgcn_wave_barrier();     // order Plds writes before reads (wave-local)

      // O += P V   (K dim = 64 keys)
      #pragma unroll
      for (int s = 0; s < 2; ++s) {
        const _Float16* pp = &Plds[wave][r][s*32];
        v16h pfr = make_frag(pp + 8*halfw, pp + 16 + 8*halfw);
        #pragma unroll
        for (int t = 0; t < 4; ++t) {
          const _Float16* vp = &Vlds[t*16 + r][s*32 + 8*halfw];
          os[t] = wmma_f16(pfr, make_frag(vp, vp + 16), os[t]);
        }
      }
    }

    // Fold this scale: ofin += (w / l) * O
    #pragma unroll
    for (int g = 0; g < 8; ++g) {
      int row  = qrow + 8*halfw + g;
      float w  = sw[((size_t)b * S_LEN + row) * NSCALE + sc];
      float f  = w / lrow[g];
      #pragma unroll
      for (int t = 0; t < 4; ++t) ofin[t][g] += f * os[t][g];
    }
  }

  // Write attention output (B, S, H) f16 for the output projection
  #pragma unroll
  for (int g = 0; g < 8; ++g) {
    int row = qrow + 8*halfw + g;
    _Float16* op = attn + ((size_t)b * S_LEN + row) * H_DIM + h * HDIM;
    #pragma unroll
    for (int t = 0; t < 4; ++t) op[t*16 + r] = (_Float16)ofin[t][g];
  }
}

// ---------------------------------------------------------------------------
// Kernel D: out = attn(f16) @ Wout + bout  (M=4096, N=1024, K=1024), f32 out
// ---------------------------------------------------------------------------
__global__ __launch_bounds__(256) void out_proj_kernel(
    const _Float16* __restrict__ attn, const float* __restrict__ Wout,
    const float* __restrict__ bout, float* __restrict__ out)
{
  __shared__ _Float16 As[128][40];
  __shared__ _Float16 Bs[128][40];
  const int m0 = blockIdx.x * 128;
  const int n0 = blockIdx.y * 128;
  const int tid  = threadIdx.x;
  const int wave = tid >> 5, lane = tid & 31;
  const int halfw = lane >> 4, r = lane & 15;
  const int wm = wave >> 2, wn = wave & 3;

  v8f acc[4][2] = {};

  for (int k0 = 0; k0 < H_DIM; k0 += 32) {
    __syncthreads();
    // A tile: already f16 (128 x 32 halves)
    #pragma unroll
    for (int i = 0; i < 2; ++i) {
      int chunk = tid + i * 256;            // 512 chunks of 8 halves
      int row = chunk >> 2;
      int cc  = (chunk & 3) * 8;
      *(uint4*)(&As[row][cc]) =
          *(const uint4*)(attn + (size_t)(m0 + row) * H_DIM + k0 + cc);
    }
    // B tile: Wout 32x128 f32, stored transposed [n][k]
    #pragma unroll
    for (int i = 0; i < 4; ++i) {
      int chunk = tid + i * 256;
      int krow = chunk >> 5;
      int nn   = (chunk & 31) * 4;
      float4 wv = *(const float4*)(Wout + (size_t)(k0 + krow) * H_DIM + n0 + nn);
      Bs[nn+0][krow] = (_Float16)wv.x;
      Bs[nn+1][krow] = (_Float16)wv.y;
      Bs[nn+2][krow] = (_Float16)wv.z;
      Bs[nn+3][krow] = (_Float16)wv.w;
    }
    __syncthreads();

    v16h af[4], bf[2];
    #pragma unroll
    for (int mt = 0; mt < 4; ++mt) {
      const _Float16* ap = &As[wm*64 + mt*16 + r][0];
      af[mt] = make_frag(ap + 8*halfw, ap + 16 + 8*halfw);
    }
    #pragma unroll
    for (int nt = 0; nt < 2; ++nt) {
      const _Float16* bp = &Bs[wn*32 + nt*16 + r][0];
      bf[nt] = make_frag(bp + 8*halfw, bp + 16 + 8*halfw);
    }
    #pragma unroll
    for (int mt = 0; mt < 4; ++mt)
      #pragma unroll
      for (int nt = 0; nt < 2; ++nt)
        acc[mt][nt] = wmma_f16(af[mt], bf[nt], acc[mt][nt]);
  }

  #pragma unroll
  for (int nt = 0; nt < 2; ++nt) {
    int n_g = n0 + wn*32 + nt*16 + r;
    float bias = bout[n_g];
    #pragma unroll
    for (int mt = 0; mt < 4; ++mt) {
      #pragma unroll
      for (int g = 0; g < 8; ++g) {
        int m_g = m0 + wm*64 + mt*16 + 8*halfw + g;
        out[(size_t)m_g * H_DIM + n_g] = acc[mt][nt][g] + bias;
      }
    }
  }
}

// ---------------------------------------------------------------------------
extern "C" void kernel_launch(void* const* d_in, const int* in_sizes, int n_in,
                              void* d_out, int out_size, void* d_ws, size_t ws_size,
                              hipStream_t stream) {
  const float* x      = (const float*)d_in[0];
  const float* Wqkv   = (const float*)d_in[1];
  const float* bqkv   = (const float*)d_in[2];
  const float* Wout   = (const float*)d_in[3];
  const float* bout   = (const float*)d_in[4];
  const float* Wscale = (const float*)d_in[5];
  const float* bscale = (const float*)d_in[6];
  float* out = (float*)d_out;

  const size_t perhead = (size_t)BATCH * NHEAD * S_LEN * HDIM;  // 4M halves
  _Float16* qf   = (_Float16*)d_ws;
  _Float16* kf   = qf + perhead;
  _Float16* vf   = kf + perhead;
  _Float16* attn = vf + perhead;                                // (B,S,H) f16
  float*    sw   = (float*)(attn + (size_t)BATCH * S_LEN * H_DIM);

  qkv_proj_kernel<<<dim3(32, 24), 256, 0, stream>>>(x, Wqkv, bqkv, qf, kf, vf);
  scale_softmax_kernel<<<dim3(512), 256, 0, stream>>>(x, Wscale, bscale, sw);
  msattn_kernel<<<dim3(32, 16), 256, 0, stream>>>(qf, kf, vf, sw, attn);
  out_proj_kernel<<<dim3(32, 8), 256, 0, stream>>>(attn, Wout, bout, out);
}
